// Experts_75840532513007
// MI455X (gfx1250) — matrix operand order repaired and checked by reading, not compiled, not run on hardware
//
#include <hip/hip_runtime.h>
#include <stdint.h>

// MoE experts FFN: out = gelu(x @ w1 + b1) @ w2 + b2, per expert.
// E=8, C=2048, D=2048, F=8192.
// One-time fp32->bf16 convert (+ weight transpose so both GEMM operands are
// K-contiguous), then bf16 WMMA GEMMs whose steady-state loop is
// global_load_async_to_lds_b128 (ASYNCcnt) + ds_load_b128 + v_wmma only.
// Block tile 128x256, 8 waves, 64x64 per wave (4x4 WMMA tiles -> 16 ds_load
// per 16 wmma per k-step).

#define E_ 8
#define C_ 2048
#define D_ 2048
#define F_ 8192

typedef __bf16 bf16_t;
typedef __bf16 bf16x4  __attribute__((ext_vector_type(4)));
typedef __bf16 bf16x8  __attribute__((ext_vector_type(8)));
typedef __bf16 bf16x16 __attribute__((ext_vector_type(16)));
typedef float  f32x4   __attribute__((ext_vector_type(4)));
typedef float  f32x8   __attribute__((ext_vector_type(8)));

#define BM 128
#define BN 256
#define BK 32
#define LDSS 48        // LDS row stride in bf16: 96B rows (16B aligned, skewed)
#define THREADS 256    // 8 wave32 waves: 2x4 grid, 64x64 tile per wave

// ---- CDNA5 async copy: global -> LDS, 16B per lane, tracked by ASYNCcnt ----
__device__ __forceinline__ void async_b128(uint32_t lds_off, const bf16_t* gptr) {
  asm volatile("global_load_async_to_lds_b128 %0, %1, off"
               :: "v"(lds_off), "v"(gptr) : "memory");
}
__device__ __forceinline__ void wait_async0() {
  asm volatile("s_wait_asynccnt 0x0" ::: "memory");
}

// A fragment (16x32 bf16): lane l (g=l>>4, r=l&15) holds row r,
// K=[g*16, g*16+16) contiguous.
__device__ __forceinline__ bf16x16 frag_a(const bf16_t* __restrict__ rowp, int g) {
  bf16x8 lo = *(const bf16x8*)(rowp + g * 16);
  bf16x8 hi = *(const bf16x8*)(rowp + g * 16 + 8);
  bf16x16 f;
#pragma unroll
  for (int i = 0; i < 8; ++i) { f[i] = lo[i]; f[i + 8] = hi[i]; }
  return f;
}
// B fragment (32x16 bf16): lane l holds column l&15, K=[g*16, g*16+16)
// contiguous; Bs rows are N (pre-transposed weights), so 2x ds_load_b128.
__device__ __forceinline__ bf16x16 frag_b(const bf16_t* __restrict__ colp) {
  bf16x8 lo = *(const bf16x8*)(colp);
  bf16x8 hi = *(const bf16x8*)(colp + 8);
  bf16x16 f;
#pragma unroll
  for (int i = 0; i < 8; ++i) { f[i] = lo[i]; f[i + 8] = hi[i]; }
  return f;
}

__device__ __forceinline__ float gelu_tanh(float t) {
  float u = 0.7978845608028654f * (t + 0.044715f * t * t * t);
  float th;
  asm("v_tanh_f32 %0, %1" : "=v"(th) : "v"(u));  // gfx1250 TRANS op
  return 0.5f * t * (1.0f + th);
}

// Stage one 128x32 A tile (8 KB) + 256x32 B tile (16 KB) via async copies:
// 6 wave-wide async issues per wave.
__device__ __forceinline__ void stage_tiles(const bf16_t* __restrict__ Ag, int lda,
                                            const bf16_t* __restrict__ Bg, int ldb,
                                            int k0, uint32_t asb, uint32_t bsb,
                                            int tid) {
#pragma unroll
  for (int i = 0; i < 2; ++i) {
    int q = tid + i * THREADS;          // 512 chunks of 8 bf16 cover 128x32
    int row = q >> 2, col = (q & 3) << 3;
    async_b128(asb + (uint32_t)(row * LDSS + col) * 2,
               Ag + (size_t)row * lda + k0 + col);
  }
#pragma unroll
  for (int i = 0; i < 4; ++i) {
    int q = tid + i * THREADS;          // 1024 chunks cover 256x32
    int row = q >> 2, col = (q & 3) << 3;
    async_b128(bsb + (uint32_t)(row * LDSS + col) * 2,
               Bg + (size_t)row * ldb + k0 + col);
  }
}

// ---------------- bf16 WMMA GEMM: out[e] = A[e] @ Bt[e]^T + bias ----------------
// A: [E][C_][KD] bf16 row-major. Bt: [E][N][KD] bf16 (N-major, K contiguous).
// out: [E][C_][N]. N = gridDim.x * BN.
template <int KD, bool GELU, typename OutT>
__global__ __launch_bounds__(THREADS, 2)
void gemm_bf16_wmma(const bf16_t* __restrict__ A, const bf16_t* __restrict__ Bt,
                    const float* __restrict__ bias, OutT* __restrict__ out) {
  __shared__ bf16_t As[2][BM * LDSS];   // 2 x 12 KB
  __shared__ bf16_t Bs[2][BN * LDSS];   // 2 x 24 KB

  const int Nsz = gridDim.x * BN;
  const int bn = blockIdx.x, bm = blockIdx.y, e = blockIdx.z;
  const int tid = threadIdx.x;
  const int lane = tid & 31, wave = tid >> 5;
  const int wm = (wave >> 2) * 64, wn = (wave & 3) * 64;
  const int g = lane >> 4, ln = lane & 15;

  const bf16_t* Ag = A + ((size_t)e * C_ + (size_t)bm * BM) * KD;
  const bf16_t* Bg = Bt + ((size_t)e * Nsz + (size_t)bn * BN) * KD;

  // LDS byte offsets (low 32 bits of the generic address = LDS offset)
  const uint32_t asb0 = (uint32_t)(uintptr_t)&As[0][0];
  const uint32_t asb1 = (uint32_t)(uintptr_t)&As[1][0];
  const uint32_t bsb0 = (uint32_t)(uintptr_t)&Bs[0][0];
  const uint32_t bsb1 = (uint32_t)(uintptr_t)&Bs[1][0];

  f32x8 acc[4][4] = {};

  stage_tiles(Ag, KD, Bg, KD, 0, asb0, bsb0, tid);

  const int nk = KD / BK;
#pragma unroll 1
  for (int kt = 0; kt < nk; ++kt) {
    wait_async0();        // this wave's tile kt landed in LDS
    __syncthreads();      // everyone's landed; previous buffer free for reuse
    const int buf = kt & 1;
    if (kt + 1 < nk)
      stage_tiles(Ag, KD, Bg, KD, (kt + 1) * BK,
                  buf ? asb0 : asb1, buf ? bsb0 : bsb1, tid);

    bf16x16 af[4], bfr[4];
#pragma unroll
    for (int i = 0; i < 4; ++i)
      af[i] = frag_a(&As[buf][(wm + i * 16 + ln) * LDSS], g);
#pragma unroll
    for (int j = 0; j < 4; ++j)
      bfr[j] = frag_b(&Bs[buf][(wn + j * 16 + ln) * LDSS + g * 16]);
#pragma unroll
    for (int i = 0; i < 4; ++i)
#pragma unroll
      for (int j = 0; j < 4; ++j)
        acc[i][j] = __builtin_amdgcn_wmma_f32_16x16x32_bf16(
            false, af[i], false, bfr[j], (short)0, acc[i][j], false, false);
  }

  // Epilogue. C/D layout: VGPR v -> row v + 8*g, col = lane&15.
  const float* be = bias + (size_t)e * Nsz;
  OutT* oe = out + (size_t)e * C_ * Nsz;
#pragma unroll
  for (int i = 0; i < 4; ++i) {
#pragma unroll
    for (int j = 0; j < 4; ++j) {
      const int col = bn * BN + wn + j * 16 + ln;
      const float b = be[col];
      const int row0 = bm * BM + wm + i * 16 + g * 8;
#pragma unroll
      for (int v = 0; v < 8; ++v) {
        float t = acc[i][j][v] + b;
        if (GELU) t = gelu_tanh(t);
        oe[(size_t)(row0 + v) * Nsz + col] = (OutT)t;
      }
    }
  }
}

// ---------------- Preprocessing kernels ----------------
// Elementwise fp32 -> bf16 (x activations), 4 elems/thread.
__global__ void cvt_f32_bf16(const float* __restrict__ s, bf16_t* __restrict__ d,
                             long n4) {
  long i = (long)blockIdx.x * blockDim.x + threadIdx.x;
  if (i >= n4) return;
  f32x4 v = ((const f32x4*)s)[i];
  bf16x4 o;
#pragma unroll
  for (int k = 0; k < 4; ++k) o[k] = (bf16_t)v[k];
  ((bf16x4*)d)[i] = o;
}

// Tiled transpose + convert: src [E][R][Cc] f32 -> dst [E][Cc][R] bf16.
__global__ __launch_bounds__(256)
void transpose_cvt(const float* __restrict__ src, bf16_t* __restrict__ dst,
                   int R, int Cc) {
  __shared__ float t[32][33];
  const int e = blockIdx.z;
  const int r0 = blockIdx.y * 32, c0 = blockIdx.x * 32;
  const float* s = src + (size_t)e * R * Cc;
  bf16_t* d = dst + (size_t)e * Cc * R;
  const int tx = threadIdx.x & 31, ty = threadIdx.x >> 5;  // 8 rows per pass
#pragma unroll
  for (int p = 0; p < 4; ++p) {
    int r = ty + p * 8;
    t[r][tx] = s[(size_t)(r0 + r) * Cc + c0 + tx];
  }
  __syncthreads();
#pragma unroll
  for (int p = 0; p < 4; ++p) {
    int c = ty + p * 8;
    d[(size_t)(c0 + c) * R + r0 + tx] = (bf16_t)t[tx][c];
  }
}

extern "C" void kernel_launch(void* const* d_in, const int* in_sizes, int n_in,
                              void* d_out, int out_size, void* d_ws, size_t ws_size,
                              hipStream_t stream) {
  const float* x  = (const float*)d_in[0];  // [1, E*C, D]
  const float* w1 = (const float*)d_in[1];  // [E, D, F]
  const float* b1 = (const float*)d_in[2];  // [E, F]
  const float* w2 = (const float*)d_in[3];  // [E, F, D]
  const float* b2 = (const float*)d_in[4];  // [E, D]
  float* out = (float*)d_out;               // [1, E*C, D]

  // ws layout (bf16): xb[E*C*D] | w1t[E*F*D] | w2t[E*D*F] | h[E*C*F] = 832 MiB
  char* ws = (char*)d_ws;
  bf16_t* xb  = (bf16_t*)ws;
  bf16_t* w1t = (bf16_t*)(ws + (size_t)E_ * C_ * D_ * 2);
  bf16_t* w2t = (bf16_t*)(ws + (size_t)E_ * C_ * D_ * 2 + (size_t)E_ * F_ * D_ * 2);
  bf16_t* hb  = (bf16_t*)(ws + (size_t)E_ * C_ * D_ * 2 + (size_t)E_ * F_ * D_ * 4);

  // 1) x -> bf16
  long n4 = (long)E_ * C_ * D_ / 4;
  cvt_f32_bf16<<<dim3((unsigned)((n4 + 255) / 256)), 256, 0, stream>>>(x, xb, n4);
  // 2) w1 [E][D][F] -> w1t [E][F][D] bf16 ; w2 [E][F][D] -> w2t [E][D][F] bf16
  transpose_cvt<<<dim3(F_ / 32, D_ / 32, E_), 256, 0, stream>>>(w1, w1t, D_, F_);
  transpose_cvt<<<dim3(D_ / 32, F_ / 32, E_), 256, 0, stream>>>(w2, w2t, F_, D_);

  // 3) h = gelu(xb @ w1t^T + b1)  (bf16 out)
  dim3 g1(F_ / BN, C_ / BM, E_);  // 32 x 16 x 8
  gemm_bf16_wmma<D_, true, bf16_t><<<g1, THREADS, 0, stream>>>(xb, w1t, b1, hb);
  // 4) out = h @ w2t^T + b2       (fp32 out)
  dim3 g2(D_ / BN, C_ / BM, E_);  // 8 x 16 x 8
  gemm_bf16_wmma<F_, false, float><<<g2, THREADS, 0, stream>>>(hb, w2t, b2, out);
}
